// CoreFringeSynergy_69913477644296
// MI455X (gfx1250) — matrix operand order repaired and checked by reading, compiled, not run on hardware
//
#include <hip/hip_runtime.h>
#include <cstddef>

// ---------------------------------------------------------------------------
// CoreFringeSynergy for MI455X (gfx1250, wave32, WMMA)
//
// Phase 1: three 2-hop graph propagations (COO SpMM via atomic scatter-add;
//          all three feature tables (~87MB) are L2-resident in the 192MB L2,
//          so this phase is L2-bandwidth bound; float2 gathers + prefetch).
// Phase 2: per-(user,bundle-candidate) core/fringe construction (gathers,
//          tiny core-MLP, softmax, top-3) -> syn[4096,128], base score.
// Phase 3: synergy MLP as two dense GEMMs on v_wmma_f32_16x16x32_f16,
//          K fully unrolled at compile time (4 + 2 WMMA ops), one wave per
//          16x16 tile so EXEC is all-ones as the WMMA spec requires.
// Phase 4: final dot + BPR (softplus) reduction to the scalar output.
// ---------------------------------------------------------------------------

#define DD      64
#define NU_     50000
#define NB_     20000
#define NI_     100000
#define N_UB_   70000
#define N_UI_   150000
#define N_BI_   120000
#define E_UB_   800000
#define E_UI_   1000000
#define E_BI_   1200000
#define BATCH_  2048
#define NPAIR_  4096      // BATCH * NCAND
#define MAXDEG_ 50

typedef __attribute__((ext_vector_type(16))) _Float16 v16h;
typedef __attribute__((ext_vector_type(8)))  float    v8f;

// -------------------------------- utility kernels --------------------------

__global__ void k_zero(float* __restrict__ p, int n) {
  int i = blockIdx.x * blockDim.x + threadIdx.x;
  if (i < n) p[i] = 0.0f;
}

__global__ void k_concat(const float* __restrict__ fa, const float* __restrict__ fb,
                         int na64, int n64,
                         float* __restrict__ fcur, float* __restrict__ facc) {
  int i = blockIdx.x * blockDim.x + threadIdx.x;
  if (i < n64) {
    float v = (i < na64) ? fa[i] : fb[i - na64];
    fcur[i] = v;
    facc[i] = v;
  }
}

__global__ void k_add(float* __restrict__ a, const float* __restrict__ b, int n) {
  int i = blockIdx.x * blockDim.x + threadIdx.x;
  if (i < n) a[i] += b[i];
}

__global__ void k_scale(float* __restrict__ a, int n, float s) {
  int i = blockIdx.x * blockDim.x + threadIdx.x;
  if (i < n) a[i] *= s;
}

// ------------------------------- SpMM (COO scatter) ------------------------
// One 32-lane group per edge; each lane handles 2 of the 64 feature dims
// (float2 gather = global_load_b64, two f32 atomic scatter-adds).
__global__ void k_spmm(const int* __restrict__ row, const int* __restrict__ col,
                       const float* __restrict__ val,
                       const float* __restrict__ x, float* __restrict__ y, int E) {
  int i = blockIdx.x * blockDim.x + threadIdx.x;
  if (i >= E * 32) return;
  int e    = i >> 5;
  int lane = i & 31;
  // look-ahead prefetch of a future gathered source row (global_prefetch_b8)
  int ep = e + 16;
  if (lane == 0 && ep < E)
    __builtin_prefetch(&x[(size_t)col[ep] * DD], 0, 1);
  int   r = row[e];
  int   c = col[e];
  float v = val[e];
  int   d = lane * 2;
  float2 xv = *(const float2*)(x + (size_t)c * DD + d);
  float* yr = y + (size_t)r * DD + d;
  atomicAdd(yr,     v * xv.x);
  atomicAdd(yr + 1, v * xv.y);
}

// ------------------------- scoring stage 1 (per pair) ----------------------
// 64 threads per block, one block per (user, candidate) pair.
__global__ void k_score_stage1(const float* __restrict__ UBacc,
                               const float* __restrict__ UIacc,
                               const float* __restrict__ BIacc,
                               const int* __restrict__ users,
                               const int* __restrict__ bundles,
                               const int* __restrict__ bundle_items,
                               const float* __restrict__ cw1,  // [2,32]
                               const float* __restrict__ cb1,  // [32]
                               const float* __restrict__ cw2,  // [32,1]
                               const float* __restrict__ cb2,  // [1]
                               float* __restrict__ syn,        // [NPAIR,128]
                               float* __restrict__ base)       // [NPAIR]
{
  const int n   = blockIdx.x;   // pair index
  const int b   = n >> 1;       // NCAND == 2
  const int tid = threadIdx.x;  // 0..63
  const int u   = users[b];
  const int bd  = bundles[n];

  __shared__ float u_ui[DD], b_bi[DD];
  __shared__ int   items[MAXDEG_];
  __shared__ float logit[MAXDEG_], pi[MAXDEG_];
  __shared__ int   chosen[MAXDEG_];
  __shared__ float red[DD];
  __shared__ float hcore_s[DD];
  __shared__ int   ti[3];
  __shared__ float tp[3];
  __shared__ int   fcnt;

  u_ui[tid] = UIacc[(size_t)u  * DD + tid];
  b_bi[tid] = BIacc[(size_t)bd * DD + tid];
  if (tid < MAXDEG_) {
    items[tid]  = bundle_items[(size_t)bd * MAXDEG_ + tid];
    chosen[tid] = 0;
  }
  __syncthreads();

  if (tid < MAXDEG_) {
    int it = items[tid];
    float sU = 0.f, sB = 0.f;
    if (it < NI_) {
      const float* rowU = UIacc + (size_t)(NU_ + it) * DD;
      const float* rowB = BIacc + (size_t)(NB_ + it) * DD;
      #pragma unroll 4
      for (int d = 0; d < DD; ++d) {
        sU += u_ui[d] * rowU[d];
        sB += b_bi[d] * rowB[d];
      }
    }
    // core MLP: [rUI,rBI] -> 32 relu -> 1
    float lg = cb2[0];
    #pragma unroll 4
    for (int j = 0; j < 32; ++j) {
      float h = sU * cw1[j] + sB * cw1[32 + j] + cb1[j];
      h = h > 0.f ? h : 0.f;
      lg += h * cw2[j];
    }
    logit[tid] = (it < NI_) ? lg : -1e30f;
  }
  __syncthreads();

  if (tid == 0) {
    // softmax over 50 masked logits
    float mx = -1e30f;
    for (int m = 0; m < MAXDEG_; ++m) mx = fmaxf(mx, logit[m]);
    float s = 0.f;
    for (int m = 0; m < MAXDEG_; ++m) { float e = expf(logit[m] - mx); pi[m] = e; s += e; }
    float inv = 1.f / s;
    for (int m = 0; m < MAXDEG_; ++m) pi[m] *= inv;
    // top-3 (first index wins ties, matching lax.top_k)
    float tsum = 0.f;
    for (int k = 0; k < 3; ++k) {
      int best = 0; float bv = -1.f;
      for (int m = 0; m < MAXDEG_; ++m)
        if (!chosen[m] && pi[m] > bv) { bv = pi[m]; best = m; }
      chosen[best] = 1; ti[k] = best; tp[k] = bv; tsum += bv;
    }
    float invt = 1.f / (tsum + 1e-10f);
    for (int k = 0; k < 3; ++k) tp[k] *= invt;
    int c = 0;
    for (int m = 0; m < MAXDEG_; ++m)
      if (items[m] < NI_ && !chosen[m]) ++c;
    fcnt = c;
  }
  __syncthreads();

  // h_core (weighted top-3) and h_fringe (masked mean of the rest)
  float hc = 0.f;
  #pragma unroll
  for (int k = 0; k < 3; ++k) {
    int it = items[ti[k]];
    float v = (it < NI_) ? UIacc[(size_t)(NU_ + it) * DD + tid] : 0.f;
    hc += v * tp[k];
  }
  float fs = 0.f;
  for (int m = 0; m < MAXDEG_; ++m) {
    int it = items[m];
    if (it < NI_ && !chosen[m]) fs += UIacc[(size_t)(NU_ + it) * DD + tid];
  }
  float hf = fs / fmaxf((float)fcnt, 1.f);
  hcore_s[tid] = hc;
  syn[(size_t)n * 128 + tid]      = hc;
  syn[(size_t)n * 128 + DD + tid] = hf;

  // base score: dot(u_ui, UB_b) + dot(UB_u, h_core)
  float ubu = UBacc[(size_t)u * DD + tid];
  float ubb = UBacc[(size_t)(NU_ + bd) * DD + tid];
  red[tid] = u_ui[tid] * ubb + ubu * hcore_s[tid];
  __syncthreads();
  for (int s = 32; s > 0; s >>= 1) {
    if (tid < s) red[tid] += red[tid + s];
    __syncthreads();
  }
  if (tid == 0) base[n] = red[0];
}

// ------------------------- WMMA GEMM: C = act(A*B + bias) ------------------
// One wave (32 threads) per 16x16 tile of C; A[M,K] f32 (lda=K), B[K,64] f32,
// converted to f16 fragments; f32 accumulate via v_wmma_f32_16x16x32_f16.
// K is a compile-time constant so the K-loop fully unrolls into a chain of
// back-to-back WMMAs with fragment packing scheduled underneath.
template <int K, bool RELU>
__global__ void k_wmma_gemm(const float* __restrict__ A,
                            const float* __restrict__ Bw,    // [K,64] row-major
                            const float* __restrict__ bias,  // [64]
                            float* __restrict__ Cout)        // [M,64]
{
  const int lane = threadIdx.x;    // 0..31
  const int g    = lane >> 4;      // half-wave group
  const int l    = lane & 15;
  const int rowA = blockIdx.x * 16 + l;   // A row (M) for this lane
  const int colN = blockIdx.y * 16 + l;   // B col (N) for this lane

  v8f acc = {};
  #pragma unroll
  for (int kb = 0; kb < K; kb += 32) {
    v16h af, bf;
    #pragma unroll
    for (int v = 0; v < 8; ++v) {
      // A 16x32 f16 layout: lanes 0-15: K {0-7,16-23}; lanes 16-31: K {8-15,24-31}
      int ka = kb + ((v < 4) ? (g * 8 + 2 * v) : (16 + g * 8 + 2 * (v - 4)));
      af[2 * v]     = (_Float16)A[(size_t)rowA * K + ka];
      af[2 * v + 1] = (_Float16)A[(size_t)rowA * K + ka + 1];
      // B 32x16 f16 layout: lanes 0-15: K {0-7,16-23}; lanes 16-31: K {8-15,24-31}
      int kc = kb + ((v < 4) ? (g * 8 + 2 * v) : (16 + g * 8 + 2 * (v - 4)));
      bf[2 * v]     = (_Float16)Bw[(size_t)kc * DD + colN];
      bf[2 * v + 1] = (_Float16)Bw[(size_t)(kc + 1) * DD + colN];
    }
    acc = __builtin_amdgcn_wmma_f32_16x16x32_f16(
        /*neg_a=*/false, af, /*neg_b=*/false, bf,
        /*c_mod=*/(short)0, acc, /*reuse_a=*/false, /*reuse_b=*/false);
  }
  // D layout: VGPR r -> M = r + 8*g ; N = lane&15
  #pragma unroll
  for (int r = 0; r < 8; ++r) {
    int m = blockIdx.x * 16 + r + g * 8;
    int n = blockIdx.y * 16 + l;
    float vv = acc[r] + bias[n];
    if (RELU) vv = vv > 0.f ? vv : 0.f;
    Cout[(size_t)m * DD + n] = vv;
  }
}

// ------------------------- final score + loss ------------------------------

__global__ void k_final_score(const float* __restrict__ UBacc,
                              const float* __restrict__ phi,
                              const float* __restrict__ base,
                              const int* __restrict__ users,
                              float* __restrict__ scores)
{
  int n = blockIdx.x, tid = threadIdx.x;
  int u = users[n >> 1];
  __shared__ float red[DD];
  red[tid] = UBacc[(size_t)u * DD + tid] * phi[(size_t)n * DD + tid];
  __syncthreads();
  for (int s = 32; s > 0; s >>= 1) {
    if (tid < s) red[tid] += red[tid + s];
    __syncthreads();
  }
  if (tid == 0) scores[n] = base[n] + red[0];
}

__global__ void k_loss(const float* __restrict__ scores, float* __restrict__ out) {
  __shared__ float red[256];
  float s = 0.f;
  for (int i = threadIdx.x; i < BATCH_; i += 256) {
    float x = scores[2 * i + 1] - scores[2 * i];
    float sp = (x > 0.f) ? (x + log1pf(expf(-x))) : log1pf(expf(x));
    s += sp;
  }
  red[threadIdx.x] = s;
  __syncthreads();
  for (int st = 128; st > 0; st >>= 1) {
    if (threadIdx.x < st) red[threadIdx.x] += red[threadIdx.x + st];
    __syncthreads();
  }
  if (threadIdx.x == 0) out[0] = red[0] / (float)BATCH_;
}

// ------------------------------- host driver -------------------------------

static inline int blks(int n) { return (n + 255) / 256; }

static void run_propagate(const int* row, const int* col, const float* val,
                          const float* fa, const float* fb,
                          int na, int n, int e,
                          float* acc, float* t0, float* t1, hipStream_t stream)
{
  int n64 = n * DD;
  k_concat<<<blks(n64), 256, 0, stream>>>(fa, fb, na * DD, n64, t0, acc);
  float* cur = t0; float* nxt = t1;
  for (int l = 0; l < 2; ++l) {
    k_zero<<<blks(n64), 256, 0, stream>>>(nxt, n64);
    k_spmm<<<blks(e * 32), 256, 0, stream>>>(row, col, val, cur, nxt, e);
    k_add<<<blks(n64), 256, 0, stream>>>(acc, nxt, n64);
    float* t = cur; cur = nxt; nxt = t;
  }
  k_scale<<<blks(n64), 256, 0, stream>>>(acc, n64, 1.0f / 3.0f);
}

extern "C" void kernel_launch(void* const* d_in, const int* in_sizes, int n_in,
                              void* d_out, int out_size, void* d_ws, size_t ws_size,
                              hipStream_t stream)
{
  (void)in_sizes; (void)n_in; (void)out_size; (void)ws_size;

  const float* usersF   = (const float*)d_in[0];
  const float* bundlesF = (const float*)d_in[1];
  const float* itemsF   = (const float*)d_in[2];
  const float* cw1      = (const float*)d_in[3];
  const float* cb1      = (const float*)d_in[4];
  const float* cw2      = (const float*)d_in[5];
  const float* cb2      = (const float*)d_in[6];
  const float* sw1      = (const float*)d_in[7];
  const float* sb1      = (const float*)d_in[8];
  const float* sw2      = (const float*)d_in[9];
  const float* sb2      = (const float*)d_in[10];
  const float* ub_val   = (const float*)d_in[11];
  const float* ui_val   = (const float*)d_in[12];
  const float* bi_val   = (const float*)d_in[13];
  const int*   users    = (const int*)d_in[14];
  const int*   bundles  = (const int*)d_in[15];
  const int*   ub_row   = (const int*)d_in[16];
  const int*   ub_col   = (const int*)d_in[17];
  const int*   ui_row   = (const int*)d_in[18];
  const int*   ui_col   = (const int*)d_in[19];
  const int*   bi_row   = (const int*)d_in[20];
  const int*   bi_col   = (const int*)d_in[21];
  const int*   bitems   = (const int*)d_in[22];
  float* out = (float*)d_out;

  // workspace carve-up (all offsets 256B-aligned; ~168 MB total)
  float* ws = (float*)d_ws;
  size_t o = 0;
  float* UBacc = ws + o; o += (size_t)N_UB_ * DD;
  float* UIacc = ws + o; o += (size_t)N_UI_ * DD;
  float* BIacc = ws + o; o += (size_t)N_BI_ * DD;
  float* t0    = ws + o; o += (size_t)N_UI_ * DD;
  float* t1    = ws + o; o += (size_t)N_UI_ * DD;
  float* syn   = ws + o; o += (size_t)NPAIR_ * 2 * DD;
  float* Hbuf  = ws + o; o += (size_t)NPAIR_ * DD;
  float* phi   = ws + o; o += (size_t)NPAIR_ * DD;
  float* base  = ws + o; o += (size_t)NPAIR_;
  float* sc    = ws + o; o += (size_t)NPAIR_;

  // Phase 1: graph propagations
  run_propagate(ub_row, ub_col, ub_val, usersF,   bundlesF, NU_, N_UB_, E_UB_, UBacc, t0, t1, stream);
  run_propagate(ui_row, ui_col, ui_val, usersF,   itemsF,   NU_, N_UI_, E_UI_, UIacc, t0, t1, stream);
  run_propagate(bi_row, bi_col, bi_val, bundlesF, itemsF,   NB_, N_BI_, E_BI_, BIacc, t0, t1, stream);

  // Phase 2: core/fringe construction -> syn, base
  k_score_stage1<<<NPAIR_, DD, 0, stream>>>(UBacc, UIacc, BIacc,
                                            users, bundles, bitems,
                                            cw1, cb1, cw2, cb2, syn, base);

  // Phase 3: synergy MLP on WMMA (K fully unrolled: 4 + 2 WMMAs)
  dim3 g1(NPAIR_ / 16, DD / 16);
  k_wmma_gemm<128, true ><<<g1, 32, 0, stream>>>(syn,  sw1, sb1, Hbuf);
  k_wmma_gemm<64,  false><<<g1, 32, 0, stream>>>(Hbuf, sw2, sb2, phi);

  // Phase 4: final dot + BPR loss
  k_final_score<<<NPAIR_, DD, 0, stream>>>(UBacc, phi, base, users, sc);
  k_loss<<<1, 256, 0, stream>>>(sc, out);
}
